// net_71837622993037
// MI455X (gfx1250) — matrix operand order repaired and checked by reading, not compiled, not run on hardware
//
#include <hip/hip_runtime.h>

#define NF 448          // concatenated feature width
#define H1 256          // hidden width
#define FOUT 64         // output width

typedef __attribute__((ext_vector_type(16))) __bf16 v16bf;
typedef __attribute__((ext_vector_type(8)))  float  v8f;

union Frag {
    v16bf v;
    uint4 q[2];
};

// ---------------------------------------------------------------- utilities
__global__ void zero_kernel(float* __restrict__ p, long long n) {
    long long i = (long long)blockIdx.x * blockDim.x + threadIdx.x;
    if (i < n) p[i] = 0.0f;
}

// W1 [448 x 256] f32 row-major  ->  W1T [256 x 448] bf16 (column-major view of W1)
__global__ void cvt_w1_kernel(const float* __restrict__ W1, __bf16* __restrict__ W1T) {
    int idx = blockIdx.x * blockDim.x + threadIdx.x;
    if (idx >= NF * H1) return;
    int k = idx >> 8;          // 0..447
    int n = idx & 255;         // 0..255
    W1T[(size_t)n * NF + k] = (__bf16)W1[idx];
}

// W2 [256 x 64] f32 row-major -> W2T [64 x 256] bf16
__global__ void cvt_w2_kernel(const float* __restrict__ W2, __bf16* __restrict__ W2T) {
    int idx = blockIdx.x * blockDim.x + threadIdx.x;
    if (idx >= H1 * FOUT) return;
    int k = idx >> 6;          // 0..255
    int n = idx & 63;          // 0..63
    W2T[(size_t)n * H1 + k] = (__bf16)W2[idx];
}

// ------------------------------------------------------------- vec_norm (h0)
// one wave32 per row of 64; lane handles columns lane and lane+32
__global__ void normalize_kernel(const float* __restrict__ x,
                                 float* __restrict__ feat, int N) {
    int lane = threadIdx.x & 31;
    int wv   = threadIdx.x >> 5;
    int row  = blockIdx.x * 8 + wv;
    if (row >= N) return;
    float a0 = x[(size_t)row * 64 + lane];
    float a1 = x[(size_t)row * 64 + lane + 32];
    float ss = a0 * a0 + a1 * a1;
    #pragma unroll
    for (int off = 16; off > 0; off >>= 1)
        ss += __shfl_xor(ss, off, 32);
    float scale = 1.0f / fmaxf(sqrtf(ss), 1e-12f);
    feat[(size_t)row * NF + lane]      = a0 * scale;
    feat[(size_t)row * NF + lane + 32] = a1 * scale;
}

// ----------------------------------------------------------------- SpMM
// edge-parallel scatter-add: thread -> (edge, channel)
__global__ void spmm_kernel(const int* __restrict__ rows, const int* __restrict__ cols,
                            const float* __restrict__ vals, float* feat,
                            int in_off, int out_off, int fshift, long long total) {
    long long idx = (long long)blockIdx.x * blockDim.x + threadIdx.x;
    if (idx >= total) return;
    int e = (int)(idx >> fshift);
    int c = (int)(idx & ((1 << fshift) - 1));
    float v = vals[e] * feat[(size_t)cols[e] * NF + in_off + c];
    atomicAdd(&feat[(size_t)rows[e] * NF + out_off + c], v);
}

// ------------------------------------------------- fused MLP with bf16 WMMA
// block = 256 threads = 8 waves, one 32-row M super-tile (2 WMMA M tiles).
// GEMM1: wave w owns hidden cols [32w, 32w+32): 2 N tiles x 2 M tiles
//        -> 4 accumulators, B fragments reused across both M tiles.
// GEMM2: 8 output tiles (2 M x 4 N), one per wave; K=256 in 8 steps.
__global__ void __launch_bounds__(256) mlp_wmma_kernel(
    const float* __restrict__ feat, const __bf16* __restrict__ W1T,
    const float* __restrict__ b1,   const __bf16* __restrict__ W2T,
    const float* __restrict__ b2,   float* __restrict__ out, int N) {

    __shared__ __align__(16) __bf16 ldsA[32 * NF];   // 28 KB: A tile (feat rows)
    __shared__ __align__(16) __bf16 ldsH[32 * H1];   // 16 KB: hidden tile

    const int tid   = threadIdx.x;
    const int lane  = tid & 31;
    const int wv    = tid >> 5;
    const int mBase = blockIdx.x * 32;

    // stage 32 feat rows, f32 -> bf16
    for (int i = tid; i < 32 * NF; i += 256) {
        int m = i / NF;
        int c = i - m * NF;
        int row = mBase + m;
        float v = (row < N) ? feat[(size_t)row * NF + c] : 0.0f;
        ldsA[i] = (__bf16)v;
    }
    __syncthreads();

    const int r  = lane & 15;   // row-in-tile (A) / column-in-tile (B,D)
    const int kh = lane >> 4;   // K-half selector

    // ---- GEMM1: [32x448] @ [448x256] + b1, ReLU
    v8f acc00 = {};   // m-tile 0, n-tile 2w
    v8f acc01 = {};   // m-tile 0, n-tile 2w+1
    v8f acc10 = {};   // m-tile 1, n-tile 2w
    v8f acc11 = {};   // m-tile 1, n-tile 2w+1
    const __bf16* w1b0 = W1T + (size_t)(wv * 32 + r) * NF;        // n-tile 2w
    const __bf16* w1b1 = w1b0 + (size_t)16 * NF;                  // n-tile 2w+1
    for (int k0 = 0; k0 < NF; k0 += 32) {
        Frag a0, a1, b0, b1f;
        const __bf16* ap0 = &ldsA[r * NF + k0 + kh * 8];
        a0.q[0] = *(const uint4*)ap0;            // K = k0+8*kh .. +7
        a0.q[1] = *(const uint4*)(ap0 + 16);     // K = k0+16+8*kh .. +7
        const __bf16* ap1 = ap0 + 16 * NF;       // rows 16..31
        a1.q[0] = *(const uint4*)ap1;
        a1.q[1] = *(const uint4*)(ap1 + 16);
        const __bf16* bp0 = w1b0 + k0 + kh * 8;
        b0.q[0] = *(const uint4*)bp0;
        b0.q[1] = *(const uint4*)(bp0 + 16);
        const __bf16* bp1 = w1b1 + k0 + kh * 8;
        b1f.q[0] = *(const uint4*)bp1;
        b1f.q[1] = *(const uint4*)(bp1 + 16);
        acc00 = __builtin_amdgcn_wmma_f32_16x16x32_bf16(false, a0.v, false, b0.v,
                                                        (short)0, acc00, false, false);
        acc01 = __builtin_amdgcn_wmma_f32_16x16x32_bf16(false, a0.v, false, b1f.v,
                                                        (short)0, acc01, false, false);
        acc10 = __builtin_amdgcn_wmma_f32_16x16x32_bf16(false, a1.v, false, b0.v,
                                                        (short)0, acc10, false, false);
        acc11 = __builtin_amdgcn_wmma_f32_16x16x32_bf16(false, a1.v, false, b1f.v,
                                                        (short)0, acc11, false, false);
    }
    {   // epilogue: bias + relu -> ldsH (bf16), both m-tiles
        int n0 = wv * 32 + r;
        int n1 = n0 + 16;
        float bias0 = b1[n0];
        float bias1 = b1[n1];
        #pragma unroll
        for (int i = 0; i < 8; ++i) {
            int m = kh * 8 + i;                  // D layout: m = i + 8*(lane>>4)
            ldsH[m * H1 + n0]        = (__bf16)fmaxf(acc00[i] + bias0, 0.0f);
            ldsH[m * H1 + n1]        = (__bf16)fmaxf(acc01[i] + bias1, 0.0f);
            ldsH[(m + 16) * H1 + n0] = (__bf16)fmaxf(acc10[i] + bias0, 0.0f);
            ldsH[(m + 16) * H1 + n1] = (__bf16)fmaxf(acc11[i] + bias1, 0.0f);
        }
    }
    __syncthreads();

    // ---- GEMM2: [32x256] @ [256x64] + b2 ; wave w -> (mtile = w>>2, ntile = w&3)
    {
        const int mt = wv >> 2;
        const int nt = wv & 3;
        v8f acc = {};
        const __bf16* w2b = W2T + (size_t)(nt * 16 + r) * H1;
        for (int k0 = 0; k0 < H1; k0 += 32) {
            Frag a, b;
            const __bf16* ap = &ldsH[(mt * 16 + r) * H1 + k0 + kh * 8];
            a.q[0] = *(const uint4*)ap;
            a.q[1] = *(const uint4*)(ap + 16);
            const __bf16* bp = w2b + k0 + kh * 8;
            b.q[0] = *(const uint4*)bp;
            b.q[1] = *(const uint4*)(bp + 16);
            acc = __builtin_amdgcn_wmma_f32_16x16x32_bf16(false, a.v, false, b.v,
                                                          (short)0, acc, false, false);
        }
        int ncol = nt * 16 + r;
        float bias = b2[ncol];
        #pragma unroll
        for (int i = 0; i < 8; ++i) {
            int m = kh * 8 + i;
            int row = mBase + mt * 16 + m;
            if (row < N) out[(size_t)row * FOUT + ncol] = acc[i] + bias;
        }
    }
}

// ---------------------------------------------------------------- launcher
extern "C" void kernel_launch(void* const* d_in, const int* in_sizes, int n_in,
                              void* d_out, int out_size, void* d_ws, size_t ws_size,
                              hipStream_t stream) {
    const float* x   = (const float*)d_in[0];
    const int*   k0r = (const int*)  d_in[1];
    const int*   k0c = (const int*)  d_in[2];
    const float* k0v = (const float*)d_in[3];
    const int*   k1r = (const int*)  d_in[4];
    const int*   k1c = (const int*)  d_in[5];
    const float* k1v = (const float*)d_in[6];
    const float* W1  = (const float*)d_in[7];
    const float* b1  = (const float*)d_in[8];
    const float* W2  = (const float*)d_in[9];
    const float* b2  = (const float*)d_in[10];
    float* out = (float*)d_out;

    const int N = in_sizes[0] / 64;
    const int E = in_sizes[1];

    // workspace layout
    char* ws = (char*)d_ws;
    float* feat = (float*)ws;                                    // N*448 f32
    size_t featBytes = ((size_t)N * NF * sizeof(float) + 255) & ~(size_t)255;
    __bf16* W1T = (__bf16*)(ws + featBytes);                     // 256*448 bf16
    size_t w1tBytes = ((size_t)NF * H1 * 2 + 255) & ~(size_t)255;
    __bf16* W2T = (__bf16*)(ws + featBytes + w1tBytes);          // 64*256 bf16

    long long featElems = (long long)N * NF;
    zero_kernel<<<(int)((featElems + 255) / 256), 256, 0, stream>>>(feat, featElems);
    cvt_w1_kernel<<<(NF * H1 + 255) / 256, 256, 0, stream>>>(W1, W1T);
    cvt_w2_kernel<<<(H1 * FOUT + 255) / 256, 256, 0, stream>>>(W2, W2T);
    normalize_kernel<<<(N + 7) / 8, 256, 0, stream>>>(x, feat, N);

    // hop 1: f=64 inputs at col 0, outputs cols 64 / 128
    long long t1 = (long long)E << 6;
    int g1 = (int)((t1 + 255) / 256);
    spmm_kernel<<<g1, 256, 0, stream>>>(k0r, k0c, k0v, feat, 0, 64, 6, t1);
    spmm_kernel<<<g1, 256, 0, stream>>>(k1r, k1c, k1v, feat, 0, 128, 6, t1);

    // hop 2: f=128 inputs at col 64, outputs cols 192 / 320
    long long t2 = (long long)E << 7;
    int g2 = (int)((t2 + 255) / 256);
    spmm_kernel<<<g2, 256, 0, stream>>>(k0r, k0c, k0v, feat, 64, 192, 7, t2);
    spmm_kernel<<<g2, 256, 0, stream>>>(k1r, k1c, k1v, feat, 64, 320, 7, t2);

    // fused MLP with WMMA (32-row super-tiles)
    mlp_wmma_kernel<<<(N + 31) / 32, 256, 0, stream>>>(feat, W1T, b1, W2T, b2, out, N);
}
